// NGLU_ablation_61967788146718
// MI455X (gfx1250) — matrix operand order repaired and checked by reading, not compiled, not run on hardware
//
#include <hip/hip_runtime.h>

// ---------------------------------------------------------------------------
// Types for CDNA5 WMMA (wave32): v_wmma_f32_16x16x32_bf16
// ---------------------------------------------------------------------------
typedef __attribute__((ext_vector_type(16))) __bf16 v16bf;
typedef __attribute__((ext_vector_type(8)))  float  v8f;

union Frag { v16bf v; uint4 q[2]; };

__device__ __forceinline__ unsigned short f2bf(float x) {
  unsigned int u = __float_as_uint(x);
  unsigned int r = ((u >> 16) & 1u) + 0x7FFFu;   // round-to-nearest-even
  return (unsigned short)((u + r) >> 16);
}
__device__ __forceinline__ float bf2f(unsigned short b) {
  return __uint_as_float(((unsigned int)b) << 16);
}

#define EPI_NONE 0
#define EPI_RELU 1
#define EPI_MED  2

// ---------------------------------------------------------------------------
// Generic bf16 WMMA GEMM: C[M,Nn] = A[M,K](bf16,row-major) @ Bpack + bias
// Bpack layout: [(kt*tn + nt)*32 + lane]*16 + e  (k = kt*32 + 16*(lane/16) + e)
// Each wave computes a 16 x (NT*16) output strip: one A fragment is reused by
// NT WMMAs per K-step. Tile-group guard is wave-uniform (EXEC all-ones).
// ---------------------------------------------------------------------------
template <int NT>
__global__ __launch_bounds__(256)
void k_gemm_bf16(const unsigned short* __restrict__ A,
                 const unsigned short* __restrict__ Bp,
                 const float* __restrict__ bias,
                 float* __restrict__ outF, int ldF,
                 unsigned short* __restrict__ outB, int ldB,
                 int M, int Nn, int K, int epi)
{
  const int lane = threadIdx.x;            // 0..31
  const int wv   = threadIdx.y;            // 0..7
  const int tn   = Nn >> 4;                // 16-wide tiles along N
  const int tgn  = tn / NT;                // tile groups along N
  const int groups = (M >> 4) * tgn;
  const int g = blockIdx.x * 8 + wv;       // wave-uniform
  if (g >= groups) return;
  const int mi = g / tgn, ngi = g % tgn;
  const int m0 = mi << 4, n0 = ngi * (NT * 16);
  const int half = lane >> 4, row = lane & 15;

  v8f acc[NT];
#pragma unroll
  for (int j = 0; j < NT; ++j) acc[j] = {};

  const int ksteps = K >> 5;
  const unsigned short* arow = A + (size_t)(m0 + row) * K + half * 16;
  const unsigned short* brow = Bp + (((size_t)ngi * NT) * 32 + lane) * 16;
  const size_t bstride = (size_t)tn * 32 * 16;   // elements per K-step plane

  for (int ks = 0; ks < ksteps; ++ks) {
    Frag fa;
    const uint4* pa = (const uint4*)(arow + ks * 32);
    fa.q[0] = pa[0];            // K chunk {half*16 .. +8}
    fa.q[1] = pa[1];            // K chunk {half*16+8 .. +8}
    if (ks + 1 < ksteps) {      // wave-uniform; near-cache prefetch of next A
      __builtin_prefetch(arow + (ks + 1) * 32, 0, 3);
    }
    Frag fb[NT];
    const uint4* pb = (const uint4*)(brow + (size_t)ks * bstride);
#pragma unroll
    for (int j = 0; j < NT; ++j) {          // consecutive 512-elem B tiles
      fb[j].q[0] = pb[j * 64 + 0];
      fb[j].q[1] = pb[j * 64 + 1];
    }
#pragma unroll
    for (int j = 0; j < NT; ++j) {
      acc[j] = __builtin_amdgcn_wmma_f32_16x16x32_bf16(
          false, fa.v, false, fb[j].v, (short)0, acc[j], false, false);
    }
  }

#pragma unroll
  for (int j = 0; j < NT; ++j) {
    const int col = n0 + j * 16 + row;
    const float bv = bias ? bias[col] : 0.0f;
#pragma unroll
    for (int r = 0; r < 8; ++r) {
      float v = acc[j][r] + bv;
      if (epi == EPI_RELU)      v = v > 0.f ? v : 0.f;
      else if (epi == EPI_MED)  v = (v + 1.0f) * 0.5f;
      const int gm = m0 + half * 8 + r;
      if (outF) outF[(size_t)gm * ldF + col] = v;
      if (outB) outB[(size_t)gm * ldB + col] = f2bf(v);
    }
  }
}

// ---------------------------------------------------------------------------
// B-operand packing into the WMMA bf16 B-fragment layout.
// ---------------------------------------------------------------------------
__global__ void k_pack(const float* __restrict__ W, unsigned short* __restrict__ P,
                       int K, int Nn)
{
  const int tid = blockIdx.x * 256 + threadIdx.x;
  if (tid >= K * Nn) return;
  const int e = tid & 15, L = (tid >> 4) & 31, tile = tid >> 9;
  const int tn = Nn >> 4;
  const int nt = tile % tn, kt = tile / tn;
  const int k  = kt * 32 + ((L < 16) ? e : 16 + e);
  const int nn = nt * 16 + (L & 15);
  P[tid] = f2bf(W[(size_t)k * Nn + nn]);
}

__global__ void k_pack_t(const float* __restrict__ W, unsigned short* __restrict__ P,
                         int K, int Nn)
{
  const int tid = blockIdx.x * 256 + threadIdx.x;
  if (tid >= K * Nn) return;
  const int e = tid & 15, L = (tid >> 4) & 31, tile = tid >> 9;
  const int tn = Nn >> 4;
  const int nt = tile % tn, kt = tile / tn;
  const int k  = kt * 32 + ((L < 16) ? e : 16 + e);
  const int nn = nt * 16 + (L & 15);
  P[tid] = f2bf(W[(size_t)nn * K + k]);
}

// ---------------------------------------------------------------------------
// Degree / norm and parcel edge bucketing
// ---------------------------------------------------------------------------
__global__ void k_deg(const int* __restrict__ dst, float* __restrict__ deg, int E)
{
  int e = blockIdx.x * 256 + threadIdx.x;
  if (e < E) atomicAdd(&deg[dst[e]], 1.0f);
}

__global__ void k_norm(const float* __restrict__ deg, float* __restrict__ nrm, int n)
{
  int i = blockIdx.x * 256 + threadIdx.x;
  if (i < n) nrm[i] = rsqrtf(deg[i] + 1.0f);
}

__global__ void k_count(const int* __restrict__ src, int* __restrict__ cnt, int E)
{
  int e = blockIdx.x * 256 + threadIdx.x;
  if (e < E) atomicAdd(&cnt[src[e] >> 7], 1);
}

// exclusive scan of 512 parcel counts (single block)
__global__ __launch_bounds__(512)
void k_scan(const int* __restrict__ cnt, int* __restrict__ offs, int* __restrict__ cur)
{
  __shared__ int tmp[512];
  const int t = threadIdx.x;
  const int v = cnt[t];
  tmp[t] = v; __syncthreads();
  for (int s = 1; s < 512; s <<= 1) {
    int add = (t >= s) ? tmp[t - s] : 0;
    __syncthreads();
    tmp[t] += add;
    __syncthreads();
  }
  const int excl = tmp[t] - v;
  offs[t] = excl;
  cur[t]  = excl;
  if (t == 511) offs[512] = tmp[511];
}

__global__ void k_bucket(const int* __restrict__ src, int* __restrict__ cur,
                         int* __restrict__ elist, int E)
{
  int e = blockIdx.x * 256 + threadIdx.x;
  if (e >= E) return;
  int p = src[e] >> 7;
  int pos = atomicAdd(&cur[p], 1);
  elist[pos] = e;
}

// ---------------------------------------------------------------------------
// Fused GCN aggregation for one (parcel, 64-channel chunk):
//   acc[n] = x[n]*nrm[n] + sum_{e:dst=n} x[src]*nrm[src]   (LDS ds_add_f32)
//   out_bf[n] = bf16(acc[n] * nrm[n])
// Edges are intra-parcel by construction, so everything lives in LDS.
// ---------------------------------------------------------------------------
__global__ __launch_bounds__(256)
void k_gcn_agg(const float* __restrict__ x, const float* __restrict__ nrm,
               const int* __restrict__ elist, const int* __restrict__ offs,
               const int* __restrict__ srcv, const int* __restrict__ dstv,
               unsigned short* __restrict__ obf, int C)
{
  __shared__ float acc[128 * 64];
  __shared__ float nl[128];
  const int b  = blockIdx.x;          // parcel
  const int c0 = blockIdx.y * 64;     // channel chunk base
  const int t  = threadIdx.x;         // 0..255
  const int base = b * 128;

  if (t < 128) nl[t] = nrm[base + t];
  __syncthreads();

  // self-loop init: acc[n, c] = x[n, c0+c] * nrm[n]
  for (int i = t; i < 128 * 64; i += 256) {
    const int n = i >> 6, c = i & 63;
    acc[i] = x[(size_t)(base + n) * C + c0 + c] * nl[n];
  }
  __syncthreads();

  // scatter edges of this parcel via LDS atomics (4 edges in flight)
  const int e0 = offs[b], e1 = offs[b + 1];
  const int sub = t >> 6;             // 0..3
  const int c   = t & 63;
  for (int e = e0 + sub; e < e1; e += 4) {
    const int ed = elist[e];
    const int s = srcv[ed] & 127, d = dstv[ed] & 127;
    atomicAdd(&acc[(d << 6) + c], x[(size_t)(base + s) * C + c0 + c] * nl[s]);
  }
  __syncthreads();

  // fused final scale + bf16 conversion
  for (int i = t; i < 128 * 64; i += 256) {
    const int n = i >> 6, c = i & 63;
    obf[(size_t)(base + n) * C + c0 + c] = f2bf(acc[i] * nl[n]);
  }
}

// ---------------------------------------------------------------------------
// Elementwise / pooling / similarity kernels
// ---------------------------------------------------------------------------
__global__ void k_f2bf(const float* __restrict__ x, unsigned short* __restrict__ o,
                       size_t n)
{
  size_t i = (size_t)blockIdx.x * 256 + threadIdx.x;
  if (i < n) o[i] = f2bf(x[i]);
}

__global__ __launch_bounds__(128)
void k_pool(const float* __restrict__ pnode, float* __restrict__ pool,
            unsigned short* __restrict__ poolbf)
{
  const int b = blockIdx.x, c = threadIdx.x;
  const float* base = pnode + (size_t)b * 128 * 128 + c;
  float s = 0.f;
  for (int t = 0; t < 128; ++t) s += base[(size_t)t * 128];
  s *= (1.0f / 128.0f);
  pool[b * 128 + c] = s;
  poolbf[b * 128 + c] = f2bf(s);
}

__global__ __launch_bounds__(256)
void k_fnorm(const float* __restrict__ hrs, const float* __restrict__ pool,
             float* __restrict__ F, unsigned short* __restrict__ Fbf)
{
  __shared__ float red[256];
  const int b = blockIdx.x, c = threadIdx.x;
  float v = (c < 128) ? hrs[b * 128 + c] : pool[b * 128 + (c - 128)];
  red[c] = v * v; __syncthreads();
  for (int s = 128; s > 0; s >>= 1) { if (c < s) red[c] += red[c + s]; __syncthreads(); }
  float inv = rsqrtf(red[0]);
  F[b * 256 + c]   = v * inv;
  Fbf[b * 256 + c] = f2bf(v * inv);
}

__global__ __launch_bounds__(256)
void k_adjrow(const float* __restrict__ med, const float* __restrict__ T0p,
              float* __restrict__ dinv)
{
  __shared__ float red[256];
  const int i = blockIdx.x, t = threadIdx.x;
  const float T0 = *T0p;
  float cnt = 0.f;
  for (int j = t; j < 512; j += 256)
    cnt += ((med[i * 512 + j] >= T0) || (j == i)) ? 1.f : 0.f;
  red[t] = cnt; __syncthreads();
  for (int s = 128; s > 0; s >>= 1) { if (t < s) red[t] += red[t + s]; __syncthreads(); }
  if (t == 0) dinv[i] = rsqrtf(red[0]);
}

__global__ void k_aadj(const float* __restrict__ med, const float* __restrict__ T0p,
                       const float* __restrict__ dinv, unsigned short* __restrict__ Abf)
{
  const int idx = blockIdx.x * 256 + threadIdx.x;  // 512*512
  const int i = idx >> 9, j = idx & 511;
  const float T0 = *T0p;
  float a = ((med[idx] >= T0) || (i == j)) ? dinv[i] * dinv[j] : 0.f;
  Abf[idx] = f2bf(a);
}

// el[r,h] = sum_c X[r, h*128+c] * al[h*128+c]   (X bf16, [R,384])
__global__ __launch_bounds__(128)
void k_logits(const unsigned short* __restrict__ X, const float* __restrict__ al,
              float* __restrict__ out)
{
  __shared__ float red[128];
  const int r = blockIdx.x, c = threadIdx.x;
  for (int h = 0; h < 3; ++h) {
    red[c] = bf2f(X[(size_t)r * 384 + h * 128 + c]) * al[h * 128 + c];
    __syncthreads();
    for (int s = 64; s > 0; s >>= 1) { if (c < s) red[c] += red[c + s]; __syncthreads(); }
    if (c == 0) out[r * 3 + h] = red[0];
    __syncthreads();
  }
}

// GAT poi->parcel: per-parcel softmax over 128 contiguous pois, weighted sum,
// + bias, mean over 3 heads -> bf16 into agg slice.
__global__ __launch_bounds__(128)
void k_gat_in(const float* __restrict__ el, const float* __restrict__ er,
              const unsigned short* __restrict__ hs_bf,
              const float* __restrict__ bg,
              unsigned short* __restrict__ outB, int ldB)
{
  __shared__ float alpha[3][128];
  __shared__ float red[128];
  const int b = blockIdx.x, t = threadIdx.x;
  const int n = b * 128 + t;
  for (int h = 0; h < 3; ++h) {
    float e = el[n * 3 + h] + er[b * 3 + h];
    e = e > 0.f ? e : 0.2f * e;              // leaky_relu(0.2)
    red[t] = e; __syncthreads();
    for (int s = 64; s > 0; s >>= 1) { if (t < s) red[t] = fmaxf(red[t], red[t + s]); __syncthreads(); }
    const float m = red[0]; __syncthreads();
    const float ex = __expf(e - m);
    red[t] = ex; __syncthreads();
    for (int s = 64; s > 0; s >>= 1) { if (t < s) red[t] += red[t + s]; __syncthreads(); }
    const float ssum = red[0]; __syncthreads();
    alpha[h][t] = ex / ssum;
  }
  __syncthreads();
  float out = 0.f;
  for (int h = 0; h < 3; ++h) {
    float s = 0.f;
    const unsigned short* base = hs_bf + (size_t)b * 128 * 384 + h * 128 + t;
    for (int p = 0; p < 128; ++p) s += alpha[h][p] * bf2f(base[(size_t)p * 384]);
    out += s + bg[h * 128 + t];
  }
  outB[(size_t)b * ldB + t] = f2bf(out * (1.0f / 3.0f));
}

// img2poi (exact: softmax over a single edge == 1; seg_mean of a per-parcel
// constant is that constant): mean over heads of (ph@Wg_w + bg_w).
__global__ __launch_bounds__(128)
void k_img2poi(const float* __restrict__ hsw, const float* __restrict__ bg,
               unsigned short* __restrict__ outB, int ldB)
{
  const int b = blockIdx.x, c = threadIdx.x;
  float v = 0.f;
  for (int h = 0; h < 3; ++h) v += hsw[(size_t)b * 384 + h * 128 + c] + bg[h * 128 + c];
  outB[(size_t)b * ldB + c] = f2bf(v * (1.0f / 3.0f));
}

// ---------------------------------------------------------------------------
// Host-side launcher
// ---------------------------------------------------------------------------
static inline void launch_gemm(hipStream_t s, const unsigned short* A,
                               const unsigned short* Bp, const float* bias,
                               float* outF, int ldF, unsigned short* outB, int ldB,
                               int M, int Nn, int K, int epi)
{
  dim3 blk(32, 8);
  if ((Nn & 63) == 0) {
    const int groups = (M >> 4) * (Nn >> 6);
    dim3 grd((groups + 7) / 8);
    k_gemm_bf16<4><<<grd, blk, 0, s>>>(A, Bp, bias, outF, ldF, outB, ldB, M, Nn, K, epi);
  } else {
    const int groups = (M >> 4) * (Nn >> 4);
    dim3 grd((groups + 7) / 8);
    k_gemm_bf16<1><<<grd, blk, 0, s>>>(A, Bp, bias, outF, ldF, outB, ldB, M, Nn, K, epi);
  }
}

extern "C" void kernel_launch(void* const* d_in, const int* in_sizes, int n_in,
                              void* d_out, int out_size, void* d_ws, size_t ws_size,
                              hipStream_t stream)
{
  (void)n_in; (void)out_size; (void)ws_size; (void)in_sizes;
  constexpr int N = 65536, E = 524288, B = 512, H = 128;

  // ---- inputs ----
  const float* poi_x   = (const float*)d_in[0];
  const float* img     = (const float*)d_in[1];
  const int*   srcp    = (const int*)d_in[2];
  const int*   dstp    = srcp + E;
  const float* T0p     = (const float*)d_in[4];
  const float* W1      = (const float*)d_in[5];  const float* b1   = (const float*)d_in[6];
  const float* W2      = (const float*)d_in[7];  const float* b2   = (const float*)d_in[8];
  const float* Whrs    = (const float*)d_in[9];  const float* bhrs = (const float*)d_in[10];
  const float* Wph     = (const float*)d_in[11]; const float* bph  = (const float*)d_in[12];
  const float* Wpp     = (const float*)d_in[13]; const float* bpp  = (const float*)d_in[14];
  const float* Wg_in   = (const float*)d_in[15]; const float* al_in = (const float*)d_in[16];
  const float* ar_in   = (const float*)d_in[17]; const float* bg_in = (const float*)d_in[18];
  const float* Wg_w    = (const float*)d_in[19];
  const float* bg_w    = (const float*)d_in[22];
  const float* Wh1     = (const float*)d_in[23]; const float* bh1  = (const float*)d_in[24];
  const float* Wp1     = (const float*)d_in[25]; const float* bp1  = (const float*)d_in[26];
  const float* Wfc     = (const float*)d_in[27]; const float* bfc  = (const float*)d_in[28];

  float* out_fc = (float*)d_out;            // [512,16]
  float* med    = (float*)d_out + 512 * 16; // [512,512]

  // ---- workspace bump allocator ----
  char* w = (char*)d_ws;
  size_t off = 0;
  auto alloc = [&](size_t bytes) -> void* {
    void* p = w + off;
    off = (off + bytes + 255) & ~(size_t)255;
    return p;
  };
  float* deg   = (float*)alloc((size_t)N * 4);
  float* nrm   = (float*)alloc((size_t)N * 4);
  int*   cnt   = (int*)alloc((size_t)B * 4);
  int*   offs  = (int*)alloc((size_t)(B + 1) * 4);
  int*   cur   = (int*)alloc((size_t)B * 4);
  int*   elist = (int*)alloc((size_t)E * 4);
  float* h1    = (float*)alloc((size_t)N * 128 * 4);   // 32MB
  unsigned short* xbf = (unsigned short*)alloc((size_t)N * 128 * 2);
  float* pnode = (float*)alloc((size_t)N * 128 * 4);
  unsigned short* pnode_bf = (unsigned short*)alloc((size_t)N * 128 * 2);
  unsigned short* pp_bf    = (unsigned short*)alloc((size_t)N * 128 * 2);
  unsigned short* hs_bf    = (unsigned short*)alloc((size_t)N * 384 * 2);  // 48MB
  float* el    = (float*)alloc((size_t)N * 3 * 4);
  unsigned short* img_bf = (unsigned short*)alloc((size_t)B * 2048 * 2);
  float* hrs   = (float*)alloc((size_t)B * H * 4);
  unsigned short* hrs_bf = (unsigned short*)alloc((size_t)B * H * 2);
  float* pool  = (float*)alloc((size_t)B * H * 4);
  unsigned short* pool_bf = (unsigned short*)alloc((size_t)B * H * 2);
  float* F     = (float*)alloc((size_t)B * 256 * 4);
  unsigned short* F_bf = (unsigned short*)alloc((size_t)B * 256 * 2);
  unsigned short* FpT  = (unsigned short*)alloc((size_t)256 * 512 * 2);
  float* dinv  = (float*)alloc((size_t)B * 4);
  unsigned short* Aadj = (unsigned short*)alloc((size_t)B * B * 2);
  float* hw    = (float*)alloc((size_t)B * H * 4);
  unsigned short* hw_p = (unsigned short*)alloc((size_t)B * H * 2);
  float* pw    = (float*)alloc((size_t)B * H * 4);
  unsigned short* pw_p = (unsigned short*)alloc((size_t)B * H * 2);
  unsigned short* ph_bf   = (unsigned short*)alloc((size_t)B * H * 2);
  unsigned short* hdin_bf = (unsigned short*)alloc((size_t)B * 384 * 2);
  float* er    = (float*)alloc((size_t)B * 3 * 4);
  float* hsw   = (float*)alloc((size_t)B * 384 * 4);
  unsigned short* agg_bf = (unsigned short*)alloc((size_t)B * 512 * 2);
  unsigned short* Whrs_p = (unsigned short*)alloc((size_t)2048 * 128 * 2);
  unsigned short* W1_p   = (unsigned short*)alloc((size_t)64 * 128 * 2);
  unsigned short* W2_p   = (unsigned short*)alloc((size_t)128 * 128 * 2);
  unsigned short* Wph_p  = (unsigned short*)alloc((size_t)128 * 128 * 2);
  unsigned short* Wpp_p  = (unsigned short*)alloc((size_t)128 * 128 * 2);
  unsigned short* Wh1_p  = (unsigned short*)alloc((size_t)128 * 128 * 2);
  unsigned short* Wp1_p  = (unsigned short*)alloc((size_t)128 * 128 * 2);
  unsigned short* Wgin_p = (unsigned short*)alloc((size_t)128 * 384 * 2);
  unsigned short* Wgw_p  = (unsigned short*)alloc((size_t)128 * 384 * 2);
  unsigned short* Wfc_p  = (unsigned short*)alloc((size_t)512 * 16 * 2);

  auto blks = [](size_t n) { return (unsigned)((n + 255) / 256); };

  // ---- degree / norm + parcel edge bucketing ----
  hipMemsetAsync(deg, 0, (size_t)N * 4, stream);
  hipMemsetAsync(cnt, 0, (size_t)B * 4, stream);
  k_deg<<<blks(E), 256, 0, stream>>>(dstp, deg, E);
  k_norm<<<blks(N), 256, 0, stream>>>(deg, nrm, N);
  k_count<<<blks(E), 256, 0, stream>>>(srcp, cnt, E);
  k_scan<<<1, 512, 0, stream>>>(cnt, offs, cur);
  k_bucket<<<blks(E), 256, 0, stream>>>(srcp, cur, elist, E);

  // ---- weight packs ----
  k_pack<<<blks(2048 * 128), 256, 0, stream>>>(Whrs, Whrs_p, 2048, 128);
  k_pack<<<blks(64 * 128),  256, 0, stream>>>(W1,  W1_p,  64, 128);
  k_pack<<<blks(128 * 128), 256, 0, stream>>>(W2,  W2_p,  128, 128);
  k_pack<<<blks(128 * 128), 256, 0, stream>>>(Wph, Wph_p, 128, 128);
  k_pack<<<blks(128 * 128), 256, 0, stream>>>(Wpp, Wpp_p, 128, 128);
  k_pack<<<blks(128 * 128), 256, 0, stream>>>(Wh1, Wh1_p, 128, 128);
  k_pack<<<blks(128 * 128), 256, 0, stream>>>(Wp1, Wp1_p, 128, 128);
  k_pack<<<blks(128 * 384), 256, 0, stream>>>(Wg_in, Wgin_p, 128, 384);
  k_pack<<<blks(128 * 384), 256, 0, stream>>>(Wg_w,  Wgw_p,  128, 384);
  k_pack<<<blks(512 * 16),  256, 0, stream>>>(Wfc, Wfc_p, 512, 16);

  // ---- hrs = img @ Whrs + bhrs ----
  k_f2bf<<<blks((size_t)B * 2048), 256, 0, stream>>>(img, img_bf, (size_t)B * 2048);
  launch_gemm(stream, img_bf, Whrs_p, bhrs, hrs, H, hrs_bf, H, B, H, 2048, EPI_NONE);

  // ---- GCN layer 1 (fused LDS aggregation -> bf16) ----
  k_gcn_agg<<<dim3(B, 1), 256, 0, stream>>>(poi_x, nrm, elist, offs, srcp, dstp, xbf, 64);
  launch_gemm(stream, xbf, W1_p, b1, h1, H, nullptr, 0, N, H, 64, EPI_RELU);

  // ---- GCN layer 2 ----
  k_gcn_agg<<<dim3(B, 2), 256, 0, stream>>>(h1, nrm, elist, offs, srcp, dstp, xbf, 128);
  launch_gemm(stream, xbf, W2_p, b2, pnode, H, pnode_bf, H, N, H, 128, EPI_NONE);

  // ---- pooling + parcel similarity (med written straight to d_out) ----
  k_pool<<<B, 128, 0, stream>>>(pnode, pool, pool_bf);
  k_fnorm<<<B, 256, 0, stream>>>(hrs, pool, F, F_bf);
  k_pack_t<<<blks(256 * 512), 256, 0, stream>>>(F, FpT, 256, 512);
  launch_gemm(stream, F_bf, FpT, nullptr, med, 512, nullptr, 0, B, 512, 256, EPI_MED);

  // ---- UniSimGraph: A = dinv * adj * dinv ----
  k_adjrow<<<B, 256, 0, stream>>>(med, T0p, dinv);
  k_aadj<<<blks((size_t)B * B), 256, 0, stream>>>(med, T0p, dinv, Aadj);

  // hrs_agg = A @ (hrs@Wh1) + bh1 -> agg_bf[:, 384:512]
  launch_gemm(stream, hrs_bf, Wh1_p, nullptr, hw, H, nullptr, 0, B, H, 128, EPI_NONE);
  k_pack<<<blks(512 * 128), 256, 0, stream>>>(hw, hw_p, 512, 128);
  launch_gemm(stream, Aadj, hw_p, bh1, nullptr, 0, agg_bf + 3 * H, 512, B, H, 512, EPI_NONE);

  // poi_agg = A @ (pool@Wp1) + bp1 -> agg_bf[:, 128:256]
  launch_gemm(stream, pool_bf, Wp1_p, nullptr, pw, H, nullptr, 0, B, H, 128, EPI_NONE);
  k_pack<<<blks(512 * 128), 256, 0, stream>>>(pw, pw_p, 512, 128);
  launch_gemm(stream, Aadj, pw_p, bp1, nullptr, 0, agg_bf + H, 512, B, H, 512, EPI_NONE);

  // ---- projections ----
  launch_gemm(stream, hrs_bf, Wph_p, bph, nullptr, 0, ph_bf, H, B, H, 128, EPI_NONE);
  launch_gemm(stream, pnode_bf, Wpp_p, bpp, nullptr, 0, pp_bf, H, N, H, 128, EPI_NONE);

  // ---- GAT poi -> parcel ----
  launch_gemm(stream, pp_bf, Wgin_p, nullptr, nullptr, 0, hs_bf, 384, N, 384, 128, EPI_NONE);
  launch_gemm(stream, ph_bf, Wgin_p, nullptr, nullptr, 0, hdin_bf, 384, B, 384, 128, EPI_NONE);
  k_logits<<<N, 128, 0, stream>>>(hs_bf, al_in, el);
  k_logits<<<B, 128, 0, stream>>>(hdin_bf, ar_in, er);
  k_gat_in<<<B, 128, 0, stream>>>(el, er, hs_bf, bg_in, agg_bf + 2 * H, 512);

  // ---- GAT parcel -> poi (exactly simplified; softmax over 1 edge == 1) ----
  launch_gemm(stream, ph_bf, Wgw_p, nullptr, hsw, 384, nullptr, 0, B, 384, 128, EPI_NONE);
  k_img2poi<<<B, 128, 0, stream>>>(hsw, bg_w, agg_bf + 0, 512);

  // ---- final FC: out = agg @ Wfc + bfc ----
  launch_gemm(stream, agg_bf, Wfc_p, bfc, out_fc, 16, nullptr, 0, B, 16, 512, EPI_NONE);
}
